// MultiQueryAttention_12463995093530
// MI455X (gfx1250) — matrix operand order repaired and checked by reading, not compiled, not run on hardware
//
#include <hip/hip_runtime.h>
#include <hip/hip_bf16.h>

typedef __bf16 bf16_t;
typedef __attribute__((ext_vector_type(16))) __bf16 v16bf;
typedef __attribute__((ext_vector_type(8)))  __bf16 bf16x8;
typedef __attribute__((ext_vector_type(8)))  float  v8f;
typedef __attribute__((ext_vector_type(4)))  unsigned int v4u;
typedef __attribute__((ext_vector_type(8)))  int v8i;
typedef __attribute__((ext_vector_type(4)))  int v4i;

#define N_EMBED 2048
#define N_HEAD  16
#define HEAD_DIM 128
#define BATCH   2
#define SEQ     2048
#define QKV_N   ((N_HEAD + 2) * HEAD_DIM)   // 2304
#define MROWS   (BATCH * SEQ)               // 4096

// ---------------------------------------------------------------------------
// f32 -> bf16 elementwise convert
// ---------------------------------------------------------------------------
__global__ void f32_to_bf16_kernel(const float* __restrict__ src,
                                   bf16_t* __restrict__ dst, int n) {
    int i = blockIdx.x * blockDim.x + threadIdx.x;
    if (i < n) dst[i] = (bf16_t)src[i];
}

// ---------------------------------------------------------------------------
// Tensor Data Mover: 2D tile (tile_rows x tile_d0 bf16) global -> LDS with
// hardware LDS row padding (pad_interval v -> pad after 2^(v+1) DWORDs,
// pad_amount v -> insert (v+1) DWORDs).  D# per CDNA5 ISA
// §8.3/8.4; groups 2/3 zero (2D tensor, fully in-bounds tiles).  TENSORcnt.
// ---------------------------------------------------------------------------
__device__ __forceinline__ void tdm_load_tile_2d(unsigned lds_addr,
                                                 const bf16_t* gptr,
                                                 unsigned tile_d0,
                                                 unsigned tile_rows,
                                                 unsigned row_stride_elems,
                                                 unsigned tensor_d0,
                                                 unsigned tensor_d1,
                                                 unsigned pad_interval,
                                                 unsigned pad_amount) {
    unsigned long long ga = (unsigned long long)(uintptr_t)gptr;
    v4u g0;
    g0.x = 1u;                                            // count=1 (valid D#)
    g0.y = lds_addr;                                      // lds_addr [63:32]
    g0.z = (unsigned)(ga & 0xFFFFFFFFu);                  // global_addr lo
    g0.w = (unsigned)((ga >> 32) & 0x01FFFFFFu) | (2u << 30); // addr hi | type=2
    v8i g1;
    g1[0] = (int)((1u << 16)                // data_size = 1 -> 2 bytes
                  | (1u << 20)              // pad_enable
                  | (pad_interval << 22)
                  | (pad_amount << 25));
    g1[1] = (int)((tensor_d0 & 0xFFFFu) << 16);           // tensor_dim0 lo16
    g1[2] = (int)(((tensor_d0 >> 16) & 0xFFFFu) |
                  ((tensor_d1 & 0xFFFFu) << 16));         // d0 hi | d1 lo
    g1[3] = (int)(((tensor_d1 >> 16) & 0xFFFFu) |
                  ((tile_d0 & 0xFFFFu) << 16));           // d1 hi | tile_dim0
    g1[4] = (int)(tile_rows & 0xFFFFu);                   // tile_dim1 | tile_dim2=0
    g1[5] = (int)row_stride_elems;                        // tensor_dim0_stride lo32
    g1[6] = 0;                                            // stride hi | dim1_stride lo
    g1[7] = 0;
    v4i z4 = {0, 0, 0, 0};
    v8i z8 = {0, 0, 0, 0, 0, 0, 0, 0};
    __builtin_amdgcn_tensor_load_to_lds(g0, g1, z4, z4, z8, 0);
}

// ---------------------------------------------------------------------------
// C[M,N](f32) = A[M,K](bf16) * B[N,K](bf16)^T  -- both operands K-major.
// Block tile 128x128, K-step 32, double-buffered TDM staging (wave 0 drives
// the DMA via a *scalar* branch, all 8 waves compute).  Each wave owns a
// 32x64 subtile -> 8 wmma per K-step from 6 ds_load_b128-fragment pairs.
// ---------------------------------------------------------------------------
template <int N, int K>
__launch_bounds__(256)
__global__ void gemm_bf16_kernel(const bf16_t* __restrict__ A,
                                 const bf16_t* __restrict__ B,
                                 float* __restrict__ C, int M) {
    constexpr int LDT = 40;                 // padded LDS row stride (elements)
    constexpr int S   = K / 32;             // K-steps
    __shared__ __align__(16) bf16_t Asmem[2][128 * LDT];
    __shared__ __align__(16) bf16_t Bsmem[2][128 * LDT];

    const int tid  = threadIdx.x;
    const int wid  = tid >> 5;
    const int lane = tid & 31;
    const int half = lane >> 4;
    const int l16  = lane & 15;
    const int wm   = wid & 3;               // 4 row groups of 32
    const int wn   = wid >> 2;              // 2 col groups of 64
    const int bm   = blockIdx.y * 128;
    const int bn   = blockIdx.x * 128;
    // wave-uniform leader flag in an SGPR -> scalar branch around TDM ops
    // (TDM ignores EXEC, so an exec-masked lowering must never reach it)
    const bool leader = (__builtin_amdgcn_readfirstlane(wid) == 0);

    const bf16_t* Ag = A + (size_t)bm * K;
    const bf16_t* Bg = B + (size_t)bn * K;
    const unsigned ldsA[2] = {(unsigned)(uintptr_t)&Asmem[0][0],
                              (unsigned)(uintptr_t)&Asmem[1][0]};
    const unsigned ldsB[2] = {(unsigned)(uintptr_t)&Bsmem[0][0],
                              (unsigned)(uintptr_t)&Bsmem[1][0]};

    v8f acc[2][4] = {};

    if (leader) {   // prologue: stage step 0 into buffer 0
        tdm_load_tile_2d(ldsA[0], Ag, 32, 128, K, K, M, 3, 3);
        tdm_load_tile_2d(ldsB[0], Bg, 32, 128, K, K, N, 3, 3);
    }

    for (int i = 0; i < S; ++i) {
        const int cur = i & 1;
        if (leader) {
            if (i + 1 < S) {
                const int k1 = 32 * (i + 1);
                tdm_load_tile_2d(ldsA[cur ^ 1], Ag + k1, 32, 128, K, K, M, 3, 3);
                tdm_load_tile_2d(ldsB[cur ^ 1], Bg + k1, 32, 128, K, K, N, 3, 3);
                __builtin_amdgcn_s_wait_tensorcnt(2);  // step i done (in-order)
            } else {
                __builtin_amdgcn_s_wait_tensorcnt(0);
            }
        }
        __syncthreads();   // buffer `cur` visible to all waves

        const bf16_t* Abuf = &Asmem[cur][0];
        const bf16_t* Bbuf = &Bsmem[cur][0];
        v16bf af[2], bfr[4];
#pragma unroll
        for (int mt = 0; mt < 2; ++mt) {
            // A 16x32 bf16 layout: lane m=l16; K chunks {0-7,16-23}/{8-15,24-31}
            const bf16_t* p = Abuf + (wm * 32 + mt * 16 + l16) * LDT;
            ((bf16x8*)&af[mt])[0] = *(const bf16x8*)(p + 8 * half);
            ((bf16x8*)&af[mt])[1] = *(const bf16x8*)(p + 16 + 8 * half);
        }
#pragma unroll
        for (int nt = 0; nt < 4; ++nt) {
            // B 32x16 bf16 layout: lane n=l16; K chunks {0-7,16-23}/{8-15,24-31}
            const bf16_t* p = Bbuf + (wn * 64 + nt * 16 + l16) * LDT;
            ((bf16x8*)&bfr[nt])[0] = *(const bf16x8*)(p + 8 * half);
            ((bf16x8*)&bfr[nt])[1] = *(const bf16x8*)(p + 16 + 8 * half);
        }
#pragma unroll
        for (int mt = 0; mt < 2; ++mt)
#pragma unroll
            for (int nt = 0; nt < 4; ++nt)
                acc[mt][nt] = __builtin_amdgcn_wmma_f32_16x16x32_bf16(
                    false, af[mt], false, bfr[nt], (short)0, acc[mt][nt],
                    false, false);

        __syncthreads();   // all waves done reading `cur` before it is reused
    }

    // epilogue: C/D layout -> element r: row = r + 8*half, col = l16.
#pragma unroll
    for (int mt = 0; mt < 2; ++mt) {
        const int row0 = bm + wm * 32 + mt * 16 + 8 * half;
#pragma unroll
        for (int nt = 0; nt < 4; ++nt) {
            float* cp = C + (size_t)row0 * N + bn + wn * 64 + nt * 16 + l16;
#pragma unroll
            for (int r = 0; r < 8; ++r)
                cp[r * N] = acc[mt][nt][r];
        }
    }
}

// ---------------------------------------------------------------------------
// RoPE + repack: qkv f32 [B,T,2304] ->
//   q bf16 [B*H, T, 128], k bf16 [B, T, 128]  (rows K-major for fragments)
// ---------------------------------------------------------------------------
__global__ void rope_pack_kernel(const float* __restrict__ qkv,
                                 bf16_t* __restrict__ qo,
                                 bf16_t* __restrict__ ko) {
    const int total = BATCH * SEQ * 17 * 64;
    int idx = blockIdx.x * blockDim.x + threadIdx.x;
    if (idx >= total) return;
    const int j  = idx & 63;
    const int hh = (idx >> 6) % 17;          // 0..15 = q heads, 16 = k
    const int bt = (idx >> 6) / 17;          // b*SEQ + t
    const int t  = bt % SEQ;
    const int b  = bt / SEQ;

    const float LN1E4 = 9.210340371976184f;
    float freq = __expf((-4.0f * (float)j - 1.0f) * (LN1E4 / 128.0f));
    float ang = (float)t * freq;
    float s, c;
    __sincosf(ang, &s, &c);

    const float* src = qkv + (size_t)bt * QKV_N + hh * HEAD_DIM;
    float x1 = src[j], x2 = src[j + 64];
    float o1 = x1 * c - x2 * s;
    float o2 = x2 * c + x1 * s;

    if (hh < N_HEAD) {
        bf16_t* dst = qo + ((size_t)(b * N_HEAD + hh) * SEQ + t) * HEAD_DIM;
        dst[j] = (bf16_t)o1;
        dst[j + 64] = (bf16_t)o2;
    } else {
        bf16_t* dst = ko + (size_t)bt * HEAD_DIM;
        dst[j] = (bf16_t)o1;
        dst[j + 64] = (bf16_t)o2;
    }
}

// v transpose-pack: vT bf16 [B, 128, T] so PV B-fragment columns (d) are
// key-index-contiguous.
__global__ void vpack_kernel(const float* __restrict__ qkv,
                             bf16_t* __restrict__ vT) {
    const int total = BATCH * SEQ * HEAD_DIM;
    int idx = blockIdx.x * blockDim.x + threadIdx.x;
    if (idx >= total) return;
    const int d  = idx & (HEAD_DIM - 1);
    const int bt = idx >> 7;
    const int t  = bt % SEQ;
    const int b  = bt / SEQ;
    vT[((size_t)(b * HEAD_DIM + d)) * SEQ + t] =
        (bf16_t)qkv[(size_t)bt * QKV_N + N_EMBED + HEAD_DIM + d];
}

// ---------------------------------------------------------------------------
// Flash attention (no causal mask): 8 waves x one 16-row q tile, streaming
// 32-key blocks.  Shared k-tile (32x128, LDS rows padded to 272B) and
// vT-tile (128x32, rows padded to 80B) staged once per block by the TDM
// (wave 0 drives via scalar branch, double-buffered).  Per block: all 8
// score B-fragments preloaded, 8 wmma for S, online softmax, P transposed
// C->A via per-wave LDS scratch, all 8 V fragments preloaded, 8 wmma for PV.
// ---------------------------------------------------------------------------
__launch_bounds__(256)
__global__ void attn_kernel(const bf16_t* __restrict__ q,
                            const bf16_t* __restrict__ k,
                            const bf16_t* __restrict__ vT,
                            bf16_t* __restrict__ out) {
    constexpr int LDP = 40;     // P scratch row stride (elements)
    constexpr int LDK = 136;    // k-tile row stride:  128 + 8  (272B)
    constexpr int LDV = 40;     // vT-tile row stride: 32 + 8   (80B)
    constexpr int S   = SEQ / 32;
    __shared__ __align__(16) bf16_t Ksmem[2][32 * LDK];
    __shared__ __align__(16) bf16_t Vsmem[2][128 * LDV];
    __shared__ __align__(16) bf16_t Psmem[8 * 16 * LDP];

    const int tid  = threadIdx.x;
    const int wid  = tid >> 5;
    const int lane = tid & 31;
    const int half = lane >> 4;
    const int l16  = lane & 15;
    const bool leader = (__builtin_amdgcn_readfirstlane(wid) == 0);

    const int qblk = blockIdx.x % (SEQ / 128);
    const int bh   = blockIdx.x / (SEQ / 128);
    const int b    = bh >> 4;
    const int h    = bh & 15;
    const int q0   = qblk * 128 + wid * 16;

    const bf16_t* qbase = q  + ((size_t)bh * SEQ + q0) * HEAD_DIM;
    const bf16_t* kbase = k  + (size_t)b * SEQ * HEAD_DIM;
    const bf16_t* vbase = vT + (size_t)b * HEAD_DIM * SEQ;
    bf16_t* Pw = Psmem + wid * 16 * LDP;

    const unsigned ldsK[2] = {(unsigned)(uintptr_t)&Ksmem[0][0],
                              (unsigned)(uintptr_t)&Ksmem[1][0]};
    const unsigned ldsV[2] = {(unsigned)(uintptr_t)&Vsmem[0][0],
                              (unsigned)(uintptr_t)&Vsmem[1][0]};

    // q A-fragments for the 4 d-chunks, live for the whole key loop
    v16bf qfrag[4];
#pragma unroll
    for (int c = 0; c < 4; ++c) {
        const bf16_t* p = qbase + (size_t)l16 * HEAD_DIM + 32 * c;
        ((bf16x8*)&qfrag[c])[0] = *(const bf16x8*)(p + 8 * half);
        ((bf16x8*)&qfrag[c])[1] = *(const bf16x8*)(p + 16 + 8 * half);
    }

    v8f oacc[8] = {};
    float mrun[8], lrun[8];
#pragma unroll
    for (int r = 0; r < 8; ++r) { mrun[r] = -1e30f; lrun[r] = 0.0f; }
    const float scale = 0.08838834764831845f;    // 1/sqrt(128)

    if (leader) {   // prologue: stage key-block 0 into buffer 0
        // k tile: 32 rows x 128 elems; pad every 256B (interval 5) by 16B
        tdm_load_tile_2d(ldsK[0], kbase, 128, 32, HEAD_DIM,
                         HEAD_DIM, SEQ, 5, 3);
        // vT tile: 128 rows x 32 elems; pad every 64B (interval 3) by 16B
        tdm_load_tile_2d(ldsV[0], vbase, 32, 128, SEQ,
                         SEQ, HEAD_DIM, 3, 3);
    }

    for (int i = 0; i < S; ++i) {
        const int cur = i & 1;
        const int kj  = i * 32;
        if (leader) {
            if (i + 1 < S) {
                const int kn = kj + 32;
                tdm_load_tile_2d(ldsK[cur ^ 1],
                                 kbase + (size_t)kn * HEAD_DIM, 128, 32,
                                 HEAD_DIM, HEAD_DIM, SEQ, 5, 3);
                tdm_load_tile_2d(ldsV[cur ^ 1],
                                 vbase + kn, 32, 128, SEQ,
                                 SEQ, HEAD_DIM, 3, 3);
                __builtin_amdgcn_s_wait_tensorcnt(2);  // block i done (in-order)
            } else {
                __builtin_amdgcn_s_wait_tensorcnt(0);
            }
        }
        __syncthreads();   // buffers `cur` visible to all waves

        const bf16_t* Kbuf = &Ksmem[cur][0];
        const bf16_t* Vbuf = &Vsmem[cur][0];

        // ---- preload all 8 score B-fragments (clause-groupable loads) ----
        v16bf kfr[8];
#pragma unroll
        for (int t = 0; t < 2; ++t)
#pragma unroll
            for (int c = 0; c < 4; ++c) {
                const bf16_t* p =
                    Kbuf + (16 * t + l16) * LDK + 32 * c;
                ((bf16x8*)&kfr[t * 4 + c])[0] = *(const bf16x8*)(p + 8 * half);
                ((bf16x8*)&kfr[t * 4 + c])[1] = *(const bf16x8*)(p + 16 + 8 * half);
            }

        // ---- scores: two 16x16 tiles (keys kj..+15, kj+16..+31) ----
        v8f s0 = {}, s1 = {};
#pragma unroll
        for (int c = 0; c < 4; ++c) {
            s0 = __builtin_amdgcn_wmma_f32_16x16x32_bf16(
                false, qfrag[c], false, kfr[c], (short)0, s0, false, false);
            s1 = __builtin_amdgcn_wmma_f32_16x16x32_bf16(
                false, qfrag[c], false, kfr[4 + c], (short)0, s1, false, false);
        }

        // ---- online softmax (row stats via 16-lane shfl reductions) ----
        float p0v[8], p1v[8];
#pragma unroll
        for (int r = 0; r < 8; ++r) {
            float a  = s0[r] * scale;
            float bb = s1[r] * scale;
            float mb = fmaxf(a, bb);
#pragma unroll
            for (int m = 1; m < 16; m <<= 1)
                mb = fmaxf(mb, __shfl_xor(mb, m, 32));
            float nm    = fmaxf(mrun[r], mb);
            float alpha = __expf(mrun[r] - nm);
            float e0 = __expf(a - nm);
            float e1 = __expf(bb - nm);
            float rs = e0 + e1;
#pragma unroll
            for (int m = 1; m < 16; m <<= 1)
                rs += __shfl_xor(rs, m, 32);
            lrun[r] = lrun[r] * alpha + rs;
            mrun[r] = nm;
            p0v[r] = e0;
            p1v[r] = e1;
#pragma unroll
            for (int n = 0; n < 8; ++n) oacc[n][r] *= alpha;
        }

        // ---- transpose P (C-layout -> A-layout) through per-wave LDS ----
        bf16_t* pw = Pw + (8 * half) * LDP + l16;
#pragma unroll
        for (int r = 0; r < 8; ++r) {
            pw[r * LDP]      = (bf16_t)p0v[r];
            pw[r * LDP + 16] = (bf16_t)p1v[r];
        }
        v16bf pf;
        const bf16_t* pr = Pw + l16 * LDP;
        ((bf16x8*)&pf)[0] = *(const bf16x8*)(pr + 8 * half);
        ((bf16x8*)&pf)[1] = *(const bf16x8*)(pr + 16 + 8 * half);

        // ---- preload all 8 V fragments, then out += P x V ----
        v16bf vfr[8];
#pragma unroll
        for (int n = 0; n < 8; ++n) {
            const bf16_t* vp = Vbuf + (16 * n + l16) * LDV;
            ((bf16x8*)&vfr[n])[0] = *(const bf16x8*)(vp + 8 * half);
            ((bf16x8*)&vfr[n])[1] = *(const bf16x8*)(vp + 16 + 8 * half);
        }
#pragma unroll
        for (int n = 0; n < 8; ++n)
            oacc[n] = __builtin_amdgcn_wmma_f32_16x16x32_bf16(
                false, pf, false, vfr[n], (short)0, oacc[n], false, false);

        __syncthreads();   // all waves done with buffers `cur`
    }

    // ---- normalize + store (head-major within the output row) ----
#pragma unroll
    for (int r = 0; r < 8; ++r) {
        float inv = 1.0f / lrun[r];
        const int row = q0 + r + 8 * half;
        bf16_t* orow =
            out + ((size_t)(b * SEQ) + row) * N_EMBED + h * HEAD_DIM;
#pragma unroll
        for (int n = 0; n < 8; ++n)
            orow[16 * n + l16] = (bf16_t)(oacc[n][r] * inv);
    }
}

// ---------------------------------------------------------------------------
extern "C" void kernel_launch(void* const* d_in, const int* in_sizes, int n_in,
                              void* d_out, int out_size, void* d_ws,
                              size_t ws_size, hipStream_t stream) {
    const float* x      = (const float*)d_in[0];
    const float* w_attn = (const float*)d_in[1];
    const float* w_out  = (const float*)d_in[2];
    float* out = (float*)d_out;

    char* ws = (char*)d_ws;
    size_t off = 0;
    auto alloc = [&](size_t bytes) {
        void* p = ws + off;
        off += (bytes + 255) & ~(size_t)255;
        return p;
    };
    bf16_t* x_bf     = (bf16_t*)alloc((size_t)MROWS * N_EMBED * 2);
    bf16_t* wattn_bf = (bf16_t*)alloc((size_t)QKV_N * N_EMBED * 2);
    bf16_t* wout_bf  = (bf16_t*)alloc((size_t)N_EMBED * N_EMBED * 2);
    float*  qkv      = (float*) alloc((size_t)MROWS * QKV_N * 4);
    bf16_t* q_bf     = (bf16_t*)alloc((size_t)BATCH * N_HEAD * SEQ * HEAD_DIM * 2);
    bf16_t* k_bf     = (bf16_t*)alloc((size_t)BATCH * SEQ * HEAD_DIM * 2);
    bf16_t* vT_bf    = (bf16_t*)alloc((size_t)BATCH * HEAD_DIM * SEQ * 2);
    bf16_t* ao_bf    = (bf16_t*)alloc((size_t)MROWS * N_EMBED * 2);

    const int n1 = MROWS * N_EMBED;
    f32_to_bf16_kernel<<<(n1 + 255) / 256, 256, 0, stream>>>(x, x_bf, n1);
    const int n2 = QKV_N * N_EMBED;
    f32_to_bf16_kernel<<<(n2 + 255) / 256, 256, 0, stream>>>(w_attn, wattn_bf, n2);
    const int n3 = N_EMBED * N_EMBED;
    f32_to_bf16_kernel<<<(n3 + 255) / 256, 256, 0, stream>>>(w_out, wout_bf, n3);

    // qkv = x @ w_attn^T   [4096 x 2304]
    gemm_bf16_kernel<QKV_N, N_EMBED>
        <<<dim3(QKV_N / 128, MROWS / 128), 256, 0, stream>>>(
            x_bf, wattn_bf, qkv, MROWS);

    const int nr = BATCH * SEQ * 17 * 64;
    rope_pack_kernel<<<(nr + 255) / 256, 256, 0, stream>>>(qkv, q_bf, k_bf);
    const int nv = BATCH * SEQ * HEAD_DIM;
    vpack_kernel<<<(nv + 255) / 256, 256, 0, stream>>>(qkv, vT_bf);

    attn_kernel<<<BATCH * N_HEAD * (SEQ / 128), 256, 0, stream>>>(
        q_bf, k_bf, vT_bf, ao_bf);

    // final: out = attn_out @ w_out^T   [4096 x 2048]
    gemm_bf16_kernel<N_EMBED, N_EMBED>
        <<<dim3(N_EMBED / 128, MROWS / 128), 256, 0, stream>>>(
            ao_bf, wout_bf, out, MROWS);
}
